// SelfAttn_39487929319918
// MI455X (gfx1250) — compile-verified, matrix operand order checked by reading, not run on hardware
//
#include <hip/hip_runtime.h>
#include <hip/hip_bf16.h>

typedef __attribute__((ext_vector_type(16))) _Float16     v16h;
typedef __attribute__((ext_vector_type(8)))  float        v8f;
typedef __attribute__((ext_vector_type(4)))  _Float16     v4h;
typedef __attribute__((ext_vector_type(4)))  unsigned int u32x4;
typedef __attribute__((ext_vector_type(8)))  int          i32x8;
typedef __attribute__((ext_vector_type(4)))  int          i32x4;

#define C_SEQ 2048
#define BATCH 4
#define DIN   512
#define DOUT  512
#define HEADS 8
#define DKH   64
#define NTOK  (C_SEQ * BATCH)   // 8192 token rows

#define HAVE_TDM __has_builtin(__builtin_amdgcn_tensor_load_to_lds)

// ---------------- CDNA5 async/TDM helpers -----------------------------------
__device__ __forceinline__ unsigned lds_off_u32(const void* p) {
    // LDS generic pointers: low 32 bits == LDS byte offset (flat LDS aperture).
    return (unsigned)(unsigned long long)p;
}

// GVS-mode async global->LDS copy, 16B per lane. Tracked by ASYNCcnt.
__device__ __forceinline__ void async_ld_b128(const void* saddr, unsigned voff,
                                              const void* lds) {
    asm volatile("global_load_async_to_lds_b128 %0, %1, %2"
                 :: "v"(lds_off_u32(lds)), "v"(voff), "s"(saddr)
                 : "memory");
}
__device__ __forceinline__ void wait_async0() {
    asm volatile("s_wait_asynccnt 0x0" ::: "memory");
}

#if HAVE_TDM
// 2D tile DMA: global (row-major, row_stride elements of 2B) -> LDS.
__device__ __forceinline__ void tdm_load_2d(const void* gaddr, unsigned lds_byte,
                                            unsigned tensor_w, unsigned tensor_h,
                                            unsigned tile_w, unsigned tile_h,
                                            unsigned row_stride) {
    const unsigned long long ga = (unsigned long long)gaddr;
    u32x4 g0;
    g0[0] = 1u;                                               // count=1 (valid D#)
    g0[1] = lds_byte;                                         // lds_addr
    g0[2] = (unsigned)ga;                                     // global_addr lo
    g0[3] = (unsigned)((ga >> 32) & 0x01FFFFFFu) | (2u << 30); // ga hi | type=2
    i32x8 g1;
    g1[0] = (int)(1u << 16);                                  // data_size=1 (2B)
    g1[1] = (int)((tensor_w & 0xFFFFu) << 16);                // tensor_dim0 lo
    g1[2] = (int)((tensor_w >> 16) | ((tensor_h & 0xFFFFu) << 16));
    g1[3] = (int)((tensor_h >> 16) | (tile_w << 16));         // tile_dim0
    g1[4] = (int)tile_h;                                      // tile_dim1 (dim2=0)
    g1[5] = (int)row_stride;                                  // dim0_stride lo
    g1[6] = 0;
    g1[7] = 0;
    i32x4 z4 = {0, 0, 0, 0};
#if defined(__clang_major__) && (__clang_major__ >= 23)
    i32x8 z8 = {0, 0, 0, 0, 0, 0, 0, 0};
    __builtin_amdgcn_tensor_load_to_lds(g0, g1, z4, z4, z8, 0);
#else
    __builtin_amdgcn_tensor_load_to_lds(g0, g1, z4, z4, 0);
#endif
}
#endif

// ---------------- WMMA fragment helpers (16x16x32 f16, wave32) --------------
__device__ __forceinline__ v16h frag_a_ld(const _Float16* src, int ld, int lane) {
    const int m = lane & 15, hf = lane >> 4;
    v16h f;
#pragma unroll
    for (int v = 0; v < 8; ++v) {
        const int k = ((v < 4) ? (2 * v) : (16 + 2 * (v - 4))) + 8 * hf;
        f[2 * v]     = src[m * ld + k];
        f[2 * v + 1] = src[m * ld + k + 1];
    }
    return f;
}
__device__ __forceinline__ v16h frag_b_ld(const _Float16* src, int ld, int lane) {
    const int n = lane & 15, hf = lane >> 4;
    v16h f;
#pragma unroll
    for (int v = 0; v < 8; ++v) {
        const int k = ((v < 4) ? (2 * v) : (16 + 2 * (v - 4))) + 8 * hf;
        f[2 * v]     = src[k * ld + n];
        f[2 * v + 1] = src[(k + 1) * ld + n];
    }
    return f;
}
__device__ __forceinline__ v8f wmma16(v16h a, v16h b, v8f c) {
    return __builtin_amdgcn_wmma_f32_16x16x32_f16(false, a, false, b,
                                                  (short)0, c, false, false);
}
__device__ __forceinline__ v8f vzero8() {
    v8f z;
#pragma unroll
    for (int i = 0; i < 8; ++i) z[i] = 0.0f;
    return z;
}

// ---------------- Kernel 1: f32 -> f16 convert ------------------------------
__global__ void cvt_f32_f16_kernel(const float* __restrict__ src,
                                   _Float16* __restrict__ dst, int n4) {
    int i = blockIdx.x * blockDim.x + threadIdx.x;
    if (i < n4) {
        const float4 f = ((const float4*)src)[i];
        v4h o;
        o[0] = (_Float16)f.x; o[1] = (_Float16)f.y;
        o[2] = (_Float16)f.z; o[3] = (_Float16)f.w;
        ((v4h*)dst)[i] = o;
    }
}

// ---------------- Kernel 2: weight transpose + convert ----------------------
__global__ __launch_bounds__(256)
void transpose_cvt_kernel(const float* __restrict__ W, _Float16* __restrict__ Wt,
                          int rows, int cols) {
    __shared__ float tile[32][33];
    const int tx = threadIdx.x, ty = threadIdx.y;
    const int x = blockIdx.x * 32 + tx;
    const int y0 = blockIdx.y * 32;
#pragma unroll
    for (int j = 0; j < 32; j += 8)
        tile[ty + j][tx] = W[(size_t)(y0 + ty + j) * cols + x];
    __syncthreads();
    const int xo = y0 + tx;
    const int yo = blockIdx.x * 32 + ty;
#pragma unroll
    for (int j = 0; j < 32; j += 8)
        Wt[(size_t)(yo + j) * rows + xo] = (_Float16)tile[tx][ty + j];
}

// ---------------- Kernel 3/5: WMMA GEMM  C = A(MxK)*Bt(KxN)+bias ------------
// WG tile 128x128, BK=32; 8 waves as 4(M) x 2(N); wave tile 32x64 (8 WMMAs/kt).
// A tile staged by global_load_async_to_lds_b128 (ASYNCcnt); B tile by TDM
// (TENSORcnt) when available. Double-buffered, one barrier per K-step.
template <bool OUT_HALF>
__global__ __launch_bounds__(256)
void gemm_wmma_kernel(const _Float16* __restrict__ A,   // M x K
                      const _Float16* __restrict__ Bt,  // K x N
                      const float* __restrict__ bias,   // N
                      void* __restrict__ Cout, int M, int N, int K) {
    constexpr int BM = 128, BN = 128, BK = 32;
    __shared__ _Float16 As[2][BM * BK];   // 8 KB x2
    __shared__ _Float16 Bs[2][BK * BN];   // 8 KB x2
    const int tid  = threadIdx.x;
    const int lane = tid & 31;
    const int wave = tid >> 5;
    const int wm   = wave & 3;            // 4 waves in M (32 rows each)
    const int wn   = wave >> 2;           // 2 waves in N (64 cols each)
    const int m0   = blockIdx.x * BM;
    const int n0   = blockIdx.y * BN;
    const int KT   = K / BK;

    v8f acc[2][4];
#pragma unroll
    for (int am = 0; am < 2; ++am)
#pragma unroll
        for (int t = 0; t < 4; ++t) acc[am][t] = vzero8();

    auto issue_tiles = [&](int kt, int buf) {
        // A tile 128x32 halves = 512 x b128 chunks -> 2 per thread
#pragma unroll
        for (int i = 0; i < 2; ++i) {
            int cs = tid + i * 256;
            int r = cs >> 2, c8 = (cs & 3) * 8;
            unsigned voff = (unsigned)(((m0 + r) * K + kt * BK + c8) * 2);
            async_ld_b128(A, voff, &As[buf][r * BK + c8]);
        }
#if HAVE_TDM
        if (wave == 0)   // one DMA descriptor moves the whole 32x128 B tile
            tdm_load_2d(Bt + (size_t)(kt * BK) * N + n0, lds_off_u32(&Bs[buf][0]),
                        (unsigned)N, (unsigned)BK, (unsigned)BN, (unsigned)BK,
                        (unsigned)N);
#else
        // B tile 32x128 halves = 512 x b128 chunks -> 2 per thread
#pragma unroll
        for (int i = 0; i < 2; ++i) {
            int cs = tid + i * 256;
            int r = cs >> 4, c8 = (cs & 15) * 8;
            unsigned voff = (unsigned)(((kt * BK + r) * N + n0 + c8) * 2);
            async_ld_b128(Bt, voff, &Bs[buf][r * BN + c8]);
        }
#endif
    };

    issue_tiles(0, 0);
    for (int kt = 0; kt < KT; ++kt) {
        wait_async0();
#if HAVE_TDM
        if (wave == 0) __builtin_amdgcn_s_wait_tensorcnt(0);
#endif
        __syncthreads();
        if (kt + 1 < KT) issue_tiles(kt + 1, (kt + 1) & 1);

        const int cur = kt & 1;
        v16h af0 = frag_a_ld(&As[cur][(wm * 32) * BK], BK, lane);
        v16h af1 = frag_a_ld(&As[cur][(wm * 32 + 16) * BK], BK, lane);
#pragma unroll
        for (int t = 0; t < 4; ++t) {
            v16h bf = frag_b_ld(&Bs[cur][wn * 64 + t * 16], BN, lane);
            acc[0][t] = wmma16(af0, bf, acc[0][t]);
            acc[1][t] = wmma16(af1, bf, acc[1][t]);
        }
    }

    const int hf = lane >> 4;
#pragma unroll
    for (int am = 0; am < 2; ++am)
#pragma unroll
        for (int t = 0; t < 4; ++t) {
            const int col = n0 + wn * 64 + t * 16 + (lane & 15);
            const float bv = bias[col];
#pragma unroll
            for (int v = 0; v < 8; ++v) {
                const int row = m0 + wm * 32 + am * 16 + v + 8 * hf;
                const float val = acc[am][t][v] + bv;
                if (OUT_HALF)
                    ((_Float16*)Cout)[(size_t)row * N + col] = (_Float16)val;
                else
                    ((float*)Cout)[(size_t)row * N + col] = val;
            }
        }
}

// ---------------- Kernel 4: flash attention ---------------------------------
// 128 threads = 4 waves; 64 q-rows per WG. K/V tiles double-buffered through
// global_load_async_to_lds_b128 so the DMA of tile jt+1 overlaps compute of jt.
__global__ __launch_bounds__(128)
void attn_kernel(const _Float16* __restrict__ Qh, const _Float16* __restrict__ Kh,
                 const _Float16* __restrict__ Vh, _Float16* __restrict__ Oh) {
    __shared__ _Float16 Qs[64 * 64];        // 8 KB
    __shared__ _Float16 Ks[2][64 * 64];     // 16 KB
    __shared__ _Float16 Vs[2][64 * 64];     // 16 KB
    __shared__ _Float16 Ps[4][16 * 64];     // 8 KB wave-private P strips

    const int tid  = threadIdx.x;
    const int lane = tid & 31;
    const int wave = tid >> 5;
    const int hf   = lane >> 4;
    const int qt   = blockIdx.x;
    const int b    = blockIdx.y >> 3;
    const int h    = blockIdx.y & 7;

    auto issue_kv = [&](int jt, int buf) {
#pragma unroll
        for (int i = 0; i < 4; ++i) {
            int cs = tid + i * 128;
            int r = cs >> 3, c8 = (cs & 7) * 8;
            unsigned voff =
                (unsigned)((((jt * 64 + r) * BATCH + b) * DOUT + h * DKH + c8) * 2);
            async_ld_b128(Kh, voff, &Ks[buf][r * 64 + c8]);
            async_ld_b128(Vh, voff, &Vs[buf][r * 64 + c8]);
        }
    };

    // Q tile (once) + first K/V tile
#pragma unroll
    for (int i = 0; i < 4; ++i) {
        int cs = tid + i * 128;
        int r = cs >> 3, c8 = (cs & 7) * 8;
        unsigned voff =
            (unsigned)((((qt * 64 + r) * BATCH + b) * DOUT + h * DKH + c8) * 2);
        async_ld_b128(Qh, voff, &Qs[r * 64 + c8]);
    }
    issue_kv(0, 0);

    float m_row[8], l_row[8];
    v8f accO[4];
#pragma unroll
    for (int v = 0; v < 8; ++v) { m_row[v] = -1e30f; l_row[v] = 0.0f; }
#pragma unroll
    for (int t = 0; t < 4; ++t) accO[t] = vzero8();

    const int JT = C_SEQ / 64;
    for (int jt = 0; jt < JT; ++jt) {
        wait_async0();
        __syncthreads();
        if (jt + 1 < JT) issue_kv(jt + 1, (jt + 1) & 1);
        const int cur = jt & 1;

        // S = Q(16x64) * K^T: K^T fragment == A-layout read of row-major K.
        v8f s[4];
#pragma unroll
        for (int t = 0; t < 4; ++t) s[t] = vzero8();
#pragma unroll
        for (int ks = 0; ks < 2; ++ks) {
            v16h qf = frag_a_ld(&Qs[(wave * 16) * 64 + ks * 32], 64, lane);
#pragma unroll
            for (int t = 0; t < 4; ++t) {
                v16h kf = frag_a_ld(&Ks[cur][(t * 16) * 64 + ks * 32], 64, lane);
                s[t] = wmma16(qf, kf, s[t]);
            }
        }
#pragma unroll
        for (int t = 0; t < 4; ++t)
#pragma unroll
            for (int v = 0; v < 8; ++v) s[t][v] *= 0.125f;   // 1/sqrt(DK)

        // Online softmax; xor masks 1/2/4/8 reduce the 16 column lanes of a row.
#pragma unroll
        for (int v = 0; v < 8; ++v) {
            float x = fmaxf(fmaxf(s[0][v], s[1][v]), fmaxf(s[2][v], s[3][v]));
            x = fmaxf(x, __shfl_xor(x, 1));
            x = fmaxf(x, __shfl_xor(x, 2));
            x = fmaxf(x, __shfl_xor(x, 4));
            x = fmaxf(x, __shfl_xor(x, 8));
            const float mnew = fmaxf(m_row[v], x);
            const float corr = __expf(m_row[v] - mnew);
            float rsum = 0.0f;
#pragma unroll
            for (int t = 0; t < 4; ++t) {
                const float p = __expf(s[t][v] - mnew);
                s[t][v] = p;
                rsum += p;
            }
            rsum += __shfl_xor(rsum, 1);
            rsum += __shfl_xor(rsum, 2);
            rsum += __shfl_xor(rsum, 4);
            rsum += __shfl_xor(rsum, 8);
            l_row[v] = l_row[v] * corr + rsum;
            m_row[v] = mnew;
#pragma unroll
            for (int t = 0; t < 4; ++t) accO[t][v] *= corr;
        }

        // P -> wave-private LDS strip (C/D layout), reload in A layout.
#pragma unroll
        for (int t = 0; t < 4; ++t)
#pragma unroll
            for (int v = 0; v < 8; ++v)
                Ps[wave][(v + 8 * hf) * 64 + t * 16 + (lane & 15)] = (_Float16)s[t][v];
        asm volatile("s_wait_dscnt 0x0" ::: "memory");

        // O += P(16x64) * V(64x64)
#pragma unroll
        for (int ks = 0; ks < 2; ++ks) {
            v16h pf = frag_a_ld(&Ps[wave][ks * 32], 64, lane);
#pragma unroll
            for (int t = 0; t < 4; ++t) {
                v16h vf = frag_b_ld(&Vs[cur][(ks * 32) * 64 + t * 16], 64, lane);
                accO[t] = wmma16(pf, vf, accO[t]);
            }
        }
    }

#pragma unroll
    for (int t = 0; t < 4; ++t)
#pragma unroll
        for (int v = 0; v < 8; ++v) {
            const int tok = qt * 64 + wave * 16 + v + 8 * hf;
            const float val = accO[t][v] / l_row[v];
            Oh[((size_t)tok * BATCH + b) * DOUT + h * DKH + t * 16 + (lane & 15)] =
                (_Float16)val;
        }
}

// ---------------- Launcher --------------------------------------------------
extern "C" void kernel_launch(void* const* d_in, const int* in_sizes, int n_in,
                              void* d_out, int out_size, void* d_ws, size_t ws_size,
                              hipStream_t stream) {
    const float* x   = (const float*)d_in[0];
    const float* Wq  = (const float*)d_in[1];
    const float* bq  = (const float*)d_in[2];
    const float* Wk  = (const float*)d_in[3];
    const float* bk  = (const float*)d_in[4];
    const float* Wv  = (const float*)d_in[5];
    const float* bv  = (const float*)d_in[6];
    const float* Wfc = (const float*)d_in[7];
    const float* bfc = (const float*)d_in[8];

    char* ws = (char*)d_ws;
    size_t off = 0;
    auto take = [&](size_t bytes) -> char* {
        char* p = ws + off;
        off = (off + bytes + 255) & ~(size_t)255;
        return p;
    };
    const size_t tok_h = (size_t)NTOK * DOUT * sizeof(_Float16);
    const size_t w_h   = (size_t)DOUT * DIN * sizeof(_Float16);
    _Float16* xh   = (_Float16*)take(tok_h);
    _Float16* Wtq  = (_Float16*)take(w_h);
    _Float16* Wtk  = (_Float16*)take(w_h);
    _Float16* Wtv  = (_Float16*)take(w_h);
    _Float16* Wtfc = (_Float16*)take(w_h);
    _Float16* Qh   = (_Float16*)take(tok_h);
    _Float16* Kh   = (_Float16*)take(tok_h);
    _Float16* Vh   = (_Float16*)take(tok_h);
    _Float16* Ah   = (_Float16*)take(tok_h);

    const int n4 = NTOK * DIN / 4;
    cvt_f32_f16_kernel<<<(n4 + 255) / 256, 256, 0, stream>>>(x, xh, n4);

    dim3 tb(32, 8), tg(DIN / 32, DOUT / 32);
    transpose_cvt_kernel<<<tg, tb, 0, stream>>>(Wq,  Wtq,  DOUT, DIN);
    transpose_cvt_kernel<<<tg, tb, 0, stream>>>(Wk,  Wtk,  DOUT, DIN);
    transpose_cvt_kernel<<<tg, tb, 0, stream>>>(Wv,  Wtv,  DOUT, DIN);
    transpose_cvt_kernel<<<tg, tb, 0, stream>>>(Wfc, Wtfc, DOUT, DOUT);

    dim3 gg(NTOK / 128, DOUT / 128);
    gemm_wmma_kernel<true><<<gg, 256, 0, stream>>>(xh, Wtq, bq, Qh, NTOK, DOUT, DIN);
    gemm_wmma_kernel<true><<<gg, 256, 0, stream>>>(xh, Wtk, bk, Kh, NTOK, DOUT, DIN);
    gemm_wmma_kernel<true><<<gg, 256, 0, stream>>>(xh, Wtv, bv, Vh, NTOK, DOUT, DIN);

    dim3 ag(C_SEQ / 64, BATCH * HEADS);
    attn_kernel<<<ag, 128, 0, stream>>>(Qh, Kh, Vh, Ah);

    gemm_wmma_kernel<false><<<gg, 256, 0, stream>>>(Ah, Wtfc, bfc, (float*)d_out,
                                                    NTOK, DOUT, DOUT);
}
